// MambONet_30210799960267
// MI455X (gfx1250) — matrix operand order repaired and checked by reading, not compiled, not run on hardware
//
#include <hip/hip_runtime.h>
#include <math.h>

typedef __attribute__((ext_vector_type(16))) _Float16 v16h;
typedef __attribute__((ext_vector_type(8)))  float    v8f;

#define BATCH   8
#define SEQ1    4096
#define SEQ2    100
#define DM      128
#define NLAYER  4
#define DI      256
#define DSTATE  16
#define DTR     8
#define DFF     512
#define NHEAD   4
#define DHEAD   32
#define BW1     (BATCH*SEQ1)      /* 32768 */
#define NCH     (BATCH*DI)        /* 2048  */
#define NCHUNK  64
#define CLEN    64
#define LDBL    48                /* padded xproj out (40->48) */
#define DROWS   816               /* decoder rows 800 padded +16 */
#define ATTN_SCALE 0.17677669529663689f

// ---------------------------------------------------------------------------
// Generic batched-strided WMMA GEMM:  C[b,h] = act(scale*A·Wᵀ + bias + R)
// A: (M,K) row-major f16, W: (N,K) row-major f16. All strides in elements.
// Contract: buffers padded so every fragment load/store is in-bounds & 16B
// aligned (rows padded to 16, K padded to 32, leading dims multiple of 8).
// Inner loop is branch-free: each wave always runs 4 N-tiles; edge blocks
// clamp the tile index (redundant but in-bounds work), stores are guarded
// only in the epilogue. This keeps the 4 accumulators pinned in VGPRs with
// no phi-copies around the v_wmma ops.
// ---------------------------------------------------------------------------
struct GemmArgs {
  const _Float16* A; int lda; long long sAb, sAh;
  const _Float16* W; int ldw; long long sWb, sWh;
  const float* bias;
  const float* R;  int ldr;  long long sRb,  sRh;
  float* Cf;       int ldc;  long long sCb,  sCh;
  _Float16* Ch;    int ldch; long long sChb, sChh;
  _Float16* Ct;    int ldct; long long sCtb, sCth;   // transposed f16 store
  int Mtiles, Ntiles, Kp, Bb, Hh; float scale; int relu;
};

__global__ __launch_bounds__(256) void k_gemm(GemmArgs g) {
  const int lane = threadIdx.x & 31;
  const int wv   = threadIdx.x >> 5;
  const int mt   = blockIdx.x * 8 + wv;
  if (mt >= g.Mtiles) return;
  const int z  = blockIdx.z;
  const int zb = z / g.Hh, zh = z % g.Hh;
  const _Float16* A = g.A + zb * g.sAb + zh * g.sAh;
  const _Float16* W = g.W + zb * g.sWb + zh * g.sWh;
  const int sub = lane >> 4, l16 = lane & 15;
  const int arow = mt * 16 + l16;
  const int nt0 = blockIdx.y * 4;
  const int last = g.Ntiles - 1;
  // clamped (uniform, branch-free) tile indices
  const int t1 = (nt0 + 1 < last) ? nt0 + 1 : last;
  const int t2 = (nt0 + 2 < last) ? nt0 + 2 : last;
  const int t3 = (nt0 + 3 < last) ? nt0 + 3 : last;

  const _Float16* ap  = A + (size_t)arow * g.lda + sub * 16;
  const _Float16* wp0 = W + (size_t)(nt0 * 16 + l16) * g.ldw + sub * 16;
  const _Float16* wp1 = W + (size_t)(t1  * 16 + l16) * g.ldw + sub * 16;
  const _Float16* wp2 = W + (size_t)(t2  * 16 + l16) * g.ldw + sub * 16;
  const _Float16* wp3 = W + (size_t)(t3  * 16 + l16) * g.ldw + sub * 16;

  v8f acc0 = {}, acc1 = {}, acc2 = {}, acc3 = {};

  for (int k = 0; k < g.Kp; k += 32) {
    union UF { uint4 u[2]; v16h v; } a, b0, b1, b2, b3;
    const uint4* ap4 = (const uint4*)(ap + k);
    a.u[0] = ap4[0]; a.u[1] = ap4[1];          // A: K chunks {0..7},{8..15} (+sub*16)
    __builtin_prefetch(ap + k + 128, 0, 3);     // WGP-scope global_prefetch_b8 on A stream
    const uint4* p0 = (const uint4*)(wp0 + k); b0.u[0] = p0[0]; b0.u[1] = p0[1];
    const uint4* p1 = (const uint4*)(wp1 + k); b1.u[0] = p1[0]; b1.u[1] = p1[1];
    const uint4* p2 = (const uint4*)(wp2 + k); b2.u[0] = p2[0]; b2.u[1] = p2[1];
    const uint4* p3 = (const uint4*)(wp3 + k); b3.u[0] = p3[0]; b3.u[1] = p3[1];
    acc0 = __builtin_amdgcn_wmma_f32_16x16x32_f16(false, a.v, false, b0.v, (short)0, acc0, false, false);
    acc1 = __builtin_amdgcn_wmma_f32_16x16x32_f16(false, a.v, false, b1.v, (short)0, acc1, false, false);
    acc2 = __builtin_amdgcn_wmma_f32_16x16x32_f16(false, a.v, false, b2.v, (short)0, acc2, false, false);
    acc3 = __builtin_amdgcn_wmma_f32_16x16x32_f16(false, a.v, false, b3.v, (short)0, acc3, false, false);
  }

  const float*    Rp  = g.R  ? g.R  + zb * g.sRb  + zh * g.sRh  : nullptr;
  float*          Cfp = g.Cf ? g.Cf + zb * g.sCb  + zh * g.sCh  : nullptr;
  _Float16*       Chp = g.Ch ? g.Ch + zb * g.sChb + zh * g.sChh : nullptr;
  _Float16*       Ctp = g.Ct ? g.Ct + zb * g.sCtb + zh * g.sCth : nullptr;

  auto epilogue = [&](v8f acc, int jt) {
    if (jt >= g.Ntiles) return;                // OOB N-tile: drop (was clamped)
    const int n = jt * 16 + l16;
    const float bv = g.bias ? g.bias[n] : 0.f;
    float vals[8];
#pragma unroll
    for (int r = 0; r < 8; r++) {
      const int m = mt * 16 + sub * 8 + r;
      float c = acc[r] * g.scale + bv;
      if (Rp) c += Rp[(size_t)m * g.ldr + n];
      if (g.relu) c = fmaxf(c, 0.f);
      vals[r] = c;
      if (Cfp) Cfp[(size_t)m * g.ldc  + n] = c;
      if (Chp) Chp[(size_t)m * g.ldch + n] = (_Float16)c;
    }
    if (Ctp) {   // C^T: per-lane 8 contiguous cols -> one b128 store
      union { uint4 u; _Float16 h[8]; } t;
#pragma unroll
      for (int r = 0; r < 8; r++) t.h[r] = (_Float16)vals[r];
      *(uint4*)(Ctp + (size_t)n * g.ldct + mt * 16 + sub * 8) = t.u;
    }
  };
  epilogue(acc0, nt0);
  epilogue(acc1, nt0 + 1);
  epilogue(acc2, nt0 + 2);
  epilogue(acc3, nt0 + 3);
}

// ---------------------------------------------------------------------------
// f32 -> f16 with row/col zero padding (weights + activations)
// ---------------------------------------------------------------------------
__global__ void k_cvt(const float* __restrict__ src, _Float16* __restrict__ dst,
                      int N, int Np, int K, int Kp) {
  int idx = blockIdx.x * 256 + threadIdx.x;
  if (idx >= Np * Kp) return;
  int n = idx / Kp, k = idx % Kp;
  dst[idx] = (n < N && k < K) ? (_Float16)src[(size_t)n * K + k] : (_Float16)0.f;
}

__global__ void k_embed_enc(const float* __restrict__ u, const float* __restrict__ ew,
                            const float* __restrict__ eb, float* __restrict__ mem) {
  int idx = blockIdx.x * 256 + threadIdx.x;
  if (idx >= BW1 * DM) return;
  int d = idx & (DM - 1), row = idx >> 7;
  mem[idx] = u[row] * ew[d] + eb[d];
}

__global__ void k_embed_dec(const float* __restrict__ y, const float* __restrict__ dw,
                            const float* __restrict__ db, float* __restrict__ X,
                            _Float16* __restrict__ Xh) {
  int idx = blockIdx.x * 256 + threadIdx.x;
  if (idx >= BATCH * SEQ2 * DM) return;
  int d = idx & (DM - 1), row = idx >> 7, s = row % SEQ2;
  float val = (y[row] * dw[d] + db[d]) * 11.313708498984761f;  // *sqrt(128)
  float div = __expf((float)(d & ~1) * (-9.210340371976184f / 128.f));
  float ang = (float)s * div;
  val += (d & 1) ? __cosf(ang) : __sinf(ang);
  X[idx] = val; Xh[idx] = (_Float16)val;
}

__global__ void k_rmsnorm(const float* __restrict__ X, const float* __restrict__ w,
                          _Float16* __restrict__ O) {
  int r = blockIdx.x, t = threadIdx.x;
  const float* x = X + (size_t)r * DM;
  __shared__ float red[DM];
  float v = x[t];
  red[t] = v * v; __syncthreads();
  for (int o = 64; o; o >>= 1) { if (t < o) red[t] += red[t + o]; __syncthreads(); }
  float rs = rsqrtf(red[0] / (float)DM + 1e-5f);
  O[(size_t)r * DM + t] = (_Float16)(v * rs * w[t]);
}

__global__ void k_layernorm(const float* __restrict__ X, const float* __restrict__ w,
                            const float* __restrict__ b, float* __restrict__ O,
                            _Float16* __restrict__ Oh) {
  int r = blockIdx.x, t = threadIdx.x;
  const float* x = X + (size_t)r * DM;
  __shared__ float red[DM];
  float v = x[t];
  red[t] = v; __syncthreads();
  for (int o = 64; o; o >>= 1) { if (t < o) red[t] += red[t + o]; __syncthreads(); }
  float mean = red[0] / (float)DM; __syncthreads();
  float d = v - mean;
  red[t] = d * d; __syncthreads();
  for (int o = 64; o; o >>= 1) { if (t < o) red[t] += red[t + o]; __syncthreads(); }
  float inv = rsqrtf(red[0] / (float)DM + 1e-5f);
  float ov = d * inv * w[t] + b[t];
  if (O)  O[(size_t)r * DM + t] = ov;
  if (Oh) Oh[(size_t)r * DM + t] = (_Float16)ov;
}

// depthwise causal conv1d (K=4) + SiLU, from f16 xz (cols 0..255)
__global__ void k_conv(const _Float16* __restrict__ xzh, const float* __restrict__ cw,
                       const float* __restrict__ cb, _Float16* __restrict__ xch) {
  int idx = blockIdx.x * 256 + threadIdx.x;
  if (idx >= BW1 * DI) return;
  int d = idx & (DI - 1); size_t bw = idx >> 8;
  int w = (int)(bw & (SEQ1 - 1));
  float acc = cb[d];
#pragma unroll
  for (int j = 0; j < 4; j++) {
    int wi = w - 3 + j;
    if (wi >= 0) acc += (float)xzh[(bw - 3 + j) * (2 * DI) + d] * cw[d * 4 + j];
  }
  xch[idx] = (_Float16)(acc / (1.f + __expf(-acc)));
}

// delta = softplus(dt @ dt_wᵀ + dt_b)
__global__ void k_delta(const float* __restrict__ dbl, const float* __restrict__ dtw,
                        const float* __restrict__ dtb, float* __restrict__ delta) {
  int idx = blockIdx.x * 256 + threadIdx.x;
  if (idx >= BW1 * DI) return;
  int d = idx & (DI - 1); size_t bw = idx >> 8;
  const float* row = dbl + bw * LDBL;
  const float* wr  = dtw + d * DTR;
  float x = dtb[d];
#pragma unroll
  for (int r = 0; r < DTR; r++) x += row[r] * wr[r];
  delta[idx] = (x > 20.f) ? x : log1pf(__expf(x));
}

// --- chunked selective scan (linear recurrence): 3 passes -------------------
__global__ void k_scan1(const float* __restrict__ delta, const float* __restrict__ dbl,
                        const _Float16* __restrict__ xch, const float* __restrict__ Alog,
                        float* __restrict__ Pq) {
  int tid = blockIdx.x * 256 + threadIdx.x;            // NCH*NCHUNK
  int ch = tid & (NCH - 1), c = tid >> 11;
  int b = ch >> 8, d = ch & (DI - 1);
  float A[DSTATE], h[DSTATE], P[DSTATE];
#pragma unroll
  for (int s = 0; s < DSTATE; s++) { A[s] = -__expf(Alog[d * DSTATE + s]); h[s] = 0.f; P[s] = 1.f; }
  int w0 = c * CLEN;
  for (int t = 0; t < CLEN; t++) {
    size_t bw = (size_t)b * SEQ1 + w0 + t;
    float dv = delta[bw * DI + d];
    float xv = (float)xch[bw * DI + d];
    float dx = dv * xv;
    const float* Br = dbl + bw * LDBL + DTR;
#pragma unroll
    for (int s = 0; s < DSTATE; s++) {
      float dA = __expf(dv * A[s]);
      h[s] = dA * h[s] + dx * Br[s];
      P[s] *= dA;
    }
  }
  float* o = Pq + ((size_t)c * NCH + ch) * 32;
#pragma unroll
  for (int s = 0; s < DSTATE; s++) { o[s] = P[s]; o[16 + s] = h[s]; }
}

__global__ void k_scan2(const float* __restrict__ Pq, float* __restrict__ h0) {
  int ch = blockIdx.x * 256 + threadIdx.x;             // NCH
  if (ch >= NCH) return;
  float hs[DSTATE];
#pragma unroll
  for (int s = 0; s < DSTATE; s++) hs[s] = 0.f;
  for (int c = 0; c < NCHUNK; c++) {
    const float* pq = Pq + ((size_t)c * NCH + ch) * 32;
    float* ho = h0 + ((size_t)c * NCH + ch) * DSTATE;
#pragma unroll
    for (int s = 0; s < DSTATE; s++) { ho[s] = hs[s]; hs[s] = pq[s] * hs[s] + pq[16 + s]; }
  }
}

__global__ void k_scan3(const float* __restrict__ delta, const float* __restrict__ dbl,
                        const _Float16* __restrict__ xch, const _Float16* __restrict__ xzh,
                        const float* __restrict__ Alog, const float* __restrict__ Dp,
                        const float* __restrict__ h0, _Float16* __restrict__ yh) {
  int tid = blockIdx.x * 256 + threadIdx.x;
  int ch = tid & (NCH - 1), c = tid >> 11;
  int b = ch >> 8, d = ch & (DI - 1);
  float A[DSTATE], h[DSTATE];
  const float* hi = h0 + ((size_t)c * NCH + ch) * DSTATE;
#pragma unroll
  for (int s = 0; s < DSTATE; s++) { A[s] = -__expf(Alog[d * DSTATE + s]); h[s] = hi[s]; }
  float Dv = Dp[d];
  int w0 = c * CLEN;
  for (int t = 0; t < CLEN; t++) {
    size_t bw = (size_t)b * SEQ1 + w0 + t;
    float dv = delta[bw * DI + d];
    float xv = (float)xch[bw * DI + d];
    float dx = dv * xv;
    const float* Br = dbl + bw * LDBL + DTR;
    const float* Cr = dbl + bw * LDBL + DTR + DSTATE;
    float acc = 0.f;
#pragma unroll
    for (int s = 0; s < DSTATE; s++) {
      float dA = __expf(dv * A[s]);
      h[s] = dA * h[s] + dx * Br[s];
      acc += h[s] * Cr[s];
    }
    float ys = acc + Dv * xv;
    float zv = (float)xzh[bw * (2 * DI) + DI + d];
    yh[bw * DI + d] = (_Float16)(ys * (zv / (1.f + __expf(-zv))));
  }
}

// softmax over rows of scores -> f16 probs (zero-fill K padding)
__global__ void k_softmax(const float* __restrict__ S, _Float16* __restrict__ P,
                          int rows, int klen, int ldS, long long sS,
                          int ldP, long long sP, int Kp) {
  int r = blockIdx.x % rows, bz = blockIdx.x / rows;
  const float* s = S + (size_t)bz * sS + (size_t)r * ldS;
  _Float16* p = P + (size_t)bz * sP + (size_t)r * ldP;
  __shared__ float red[128];
  int t = threadIdx.x;
  float mx = -1e30f;
  for (int i = t; i < klen; i += 128) mx = fmaxf(mx, s[i]);
  red[t] = mx; __syncthreads();
  for (int o = 64; o; o >>= 1) { if (t < o) red[t] = fmaxf(red[t], red[t + o]); __syncthreads(); }
  mx = red[0]; __syncthreads();
  float sum = 0.f;
  for (int i = t; i < klen; i += 128) sum += __expf(s[i] - mx);
  red[t] = sum; __syncthreads();
  for (int o = 64; o; o >>= 1) { if (t < o) red[t] += red[t + o]; __syncthreads(); }
  float inv = 1.f / red[0];
  for (int i = t; i < klen; i += 128) p[i] = (_Float16)(__expf(s[i] - mx) * inv);
  for (int i = klen + t; i < Kp; i += 128) p[i] = (_Float16)0.f;
}

__global__ void k_extract(const float* __restrict__ xf, float* __restrict__ out) {
  int i = blockIdx.x * 256 + threadIdx.x;
  if (i >= BATCH * SEQ2) return;
  out[i] = xf[(size_t)i * DM + 0];
  out[BATCH * SEQ2 + i] = xf[(size_t)i * DM + 1];
}

// ---------------------------------------------------------------------------
static inline int cdiv(int a, int b) { return (a + b - 1) / b; }

static void launch_gemm(hipStream_t st, GemmArgs g, int M, int N) {
  g.Mtiles = cdiv(M, 16); g.Ntiles = cdiv(N, 16);
  if (g.Bb == 0) g.Bb = 1;
  if (g.Hh == 0) g.Hh = 1;
  if (g.scale == 0.f) g.scale = 1.f;
  dim3 grid(cdiv(g.Mtiles, 8), cdiv(g.Ntiles, 4), g.Bb * g.Hh);
  k_gemm<<<grid, dim3(256), 0, st>>>(g);
}

extern "C" void kernel_launch(void* const* d_in, const int* in_sizes, int n_in,
                              void* d_out, int out_size, void* d_ws, size_t ws_size,
                              hipStream_t stream) {
  const float* u        = (const float*)d_in[0];
  const float* yin      = (const float*)d_in[1];
  const float* emb_w    = (const float*)d_in[2];
  const float* emb_b    = (const float*)d_in[3];
  const float* m_norm_w = (const float*)d_in[4];
  const float* m_in_w   = (const float*)d_in[5];
  const float* m_conv_w = (const float*)d_in[6];
  const float* m_conv_b = (const float*)d_in[7];
  const float* m_xproj_w= (const float*)d_in[8];
  const float* m_dt_w   = (const float*)d_in[9];
  const float* m_dt_b   = (const float*)d_in[10];
  const float* m_Alog   = (const float*)d_in[11];
  const float* m_D      = (const float*)d_in[12];
  const float* m_out_w  = (const float*)d_in[13];
  const float* d_emb_w  = (const float*)d_in[14];
  const float* d_emb_b  = (const float*)d_in[15];
  const float* sa_in_w  = (const float*)d_in[16];
  const float* sa_in_b  = (const float*)d_in[17];
  const float* sa_out_w = (const float*)d_in[18];
  const float* sa_out_b = (const float*)d_in[19];
  const float* ca_in_w  = (const float*)d_in[20];
  const float* ca_in_b  = (const float*)d_in[21];
  const float* ca_out_w = (const float*)d_in[22];
  const float* ca_out_b = (const float*)d_in[23];
  const float* ff_w1    = (const float*)d_in[24];
  const float* ff_b1    = (const float*)d_in[25];
  const float* ff_w2    = (const float*)d_in[26];
  const float* ff_b2    = (const float*)d_in[27];
  const float* ln1_w    = (const float*)d_in[28];
  const float* ln1_b    = (const float*)d_in[29];
  const float* ln2_w    = (const float*)d_in[30];
  const float* ln2_b    = (const float*)d_in[31];
  const float* ln3_w    = (const float*)d_in[32];
  const float* ln3_b    = (const float*)d_in[33];
  const float* fn_w     = (const float*)d_in[34];
  const float* fn_b     = (const float*)d_in[35];
  float* out = (float*)d_out;
  char* ws = (char*)d_ws;

  size_t cur = 0;
  auto take = [&](size_t bytes) -> char* {
    char* p = ws + cur;
    cur += (bytes + 255) & ~(size_t)255;
    return p;
  };

  // ---- persistent ----
  float*     mem   = (float*)take((size_t)BW1 * DM * 4);
  _Float16*  memh  = (_Float16*)take((size_t)BW1 * DM * 2);
  _Float16*  inwh  = (_Float16*)take((size_t)NLAYER * 2 * DI * DM * 2);
  _Float16*  xprojh= (_Float16*)take((size_t)NLAYER * LDBL * DI * 2);
  _Float16*  outwh = (_Float16*)take((size_t)NLAYER * DM * DI * 2);
  _Float16*  sainh = (_Float16*)take((size_t)NLAYER * 3 * DM * DM * 2);
  _Float16*  saouth= (_Float16*)take((size_t)NLAYER * DM * DM * 2);
  _Float16*  cainh = (_Float16*)take((size_t)NLAYER * 3 * DM * DM * 2);
  _Float16*  caouth= (_Float16*)take((size_t)NLAYER * DM * DM * 2);
  _Float16*  ffw1h = (_Float16*)take((size_t)NLAYER * DFF * DM * 2);
  _Float16*  ffw2h = (_Float16*)take((size_t)NLAYER * DM * DFF * 2);
  float*     xdec  = (float*)take((size_t)DROWS * DM * 4);
  _Float16*  xh    = (_Float16*)take((size_t)DROWS * DM * 2);
  float*     tmp   = (float*)take((size_t)DROWS * DM * 4);
  float*     xf    = (float*)take((size_t)DROWS * DM * 4);

  size_t tbase = cur;
  // ---- encoder temps (union region) ----
  _Float16* normh = (_Float16*)take((size_t)BW1 * DM * 2);
  _Float16* xzh   = (_Float16*)take((size_t)BW1 * 2 * DI * 2);
  _Float16* xch   = (_Float16*)take((size_t)BW1 * DI * 2);
  float*    dbl   = (float*)take((size_t)BW1 * LDBL * 4);
  float*    delta = (float*)take((size_t)BW1 * DI * 4);
  float*    Pq    = (float*)take((size_t)NCHUNK * NCH * 32 * 4);
  float*    h0    = (float*)take((size_t)NCHUNK * NCH * DSTATE * 4);
  _Float16* yhbuf = (_Float16*)take((size_t)BW1 * DI * 2);
  // ---- decoder temps (overlay same region) ----
  cur = tbase;
  _Float16* khc    = (_Float16*)take((size_t)BW1 * DM * 2);
  _Float16* Vtc    = (_Float16*)take((size_t)BATCH * DM * SEQ1 * 2);
  float*    scoresC= (float*)take((size_t)BATCH * NHEAD * 112 * SEQ1 * 4);
  _Float16* probsC = (_Float16*)take((size_t)BATCH * NHEAD * 112 * SEQ1 * 2);
  _Float16* qh     = (_Float16*)take((size_t)DROWS * DM * 2);
  _Float16* kh     = (_Float16*)take((size_t)DROWS * DM * 2);
  _Float16* VtS    = (_Float16*)take((size_t)BATCH * DM * 128 * 2);
  float*    scoresS= (float*)take((size_t)BATCH * NHEAD * 112 * 112 * 4);
  _Float16* probsS = (_Float16*)take((size_t)BATCH * NHEAD * 112 * 128 * 2);
  _Float16* attnh  = (_Float16*)take((size_t)DROWS * DM * 2);
  _Float16* ffh    = (_Float16*)take((size_t)DROWS * DFF * 2);
  (void)ws_size; (void)in_sizes; (void)n_in; (void)out_size;

  auto cvt = [&](const float* s, _Float16* d, int N, int Np, int K, int Kp) {
    k_cvt<<<cdiv(Np * Kp, 256), 256, 0, stream>>>(s, d, N, Np, K, Kp);
  };

  // ---- weight f16 conversion ----
  cvt(m_in_w,  inwh,  NLAYER * 2 * DI, NLAYER * 2 * DI, DM, DM);
  for (int l = 0; l < NLAYER; l++)
    cvt(m_xproj_w + (size_t)l * 40 * DI, xprojh + (size_t)l * LDBL * DI, 40, LDBL, DI, DI);
  cvt(m_out_w, outwh, NLAYER * DM, NLAYER * DM, DI, DI);
  cvt(sa_in_w, sainh, NLAYER * 3 * DM, NLAYER * 3 * DM, DM, DM);
  cvt(sa_out_w, saouth, NLAYER * DM, NLAYER * DM, DM, DM);
  cvt(ca_in_w, cainh, NLAYER * 3 * DM, NLAYER * 3 * DM, DM, DM);
  cvt(ca_out_w, caouth, NLAYER * DM, NLAYER * DM, DM, DM);
  cvt(ff_w1, ffw1h, NLAYER * DFF, NLAYER * DFF, DM, DM);
  cvt(ff_w2, ffw2h, NLAYER * DM, NLAYER * DM, DFF, DFF);

  // ---- Mamba encoder ----
  k_embed_enc<<<cdiv(BW1 * DM, 256), 256, 0, stream>>>(u, emb_w, emb_b, mem);
  for (int l = 0; l < NLAYER; l++) {
    k_rmsnorm<<<BW1, DM, 0, stream>>>(mem, m_norm_w + l * DM, normh);
    { GemmArgs g{}; g.A = normh; g.lda = DM; g.W = inwh + (size_t)l * 2 * DI * DM; g.ldw = DM;
      g.Ch = xzh; g.ldch = 2 * DI; g.Kp = DM;
      launch_gemm(stream, g, BW1, 2 * DI); }
    k_conv<<<cdiv(BW1 * DI, 256), 256, 0, stream>>>(xzh, m_conv_w + l * DI * 4,
                                                    m_conv_b + l * DI, xch);
    { GemmArgs g{}; g.A = xch; g.lda = DI; g.W = xprojh + (size_t)l * LDBL * DI; g.ldw = DI;
      g.Cf = dbl; g.ldc = LDBL; g.Kp = DI;
      launch_gemm(stream, g, BW1, LDBL); }
    k_delta<<<cdiv(BW1 * DI, 256), 256, 0, stream>>>(dbl, m_dt_w + l * DI * DTR,
                                                     m_dt_b + l * DI, delta);
    k_scan1<<<NCH * NCHUNK / 256, 256, 0, stream>>>(delta, dbl, xch,
                                                    m_Alog + (size_t)l * DI * DSTATE, Pq);
    k_scan2<<<NCH / 256, 256, 0, stream>>>(Pq, h0);
    k_scan3<<<NCH * NCHUNK / 256, 256, 0, stream>>>(delta, dbl, xch, xzh,
                                                    m_Alog + (size_t)l * DI * DSTATE,
                                                    m_D + l * DI, h0, yhbuf);
    { GemmArgs g{}; g.A = yhbuf; g.lda = DI; g.W = outwh + (size_t)l * DM * DI; g.ldw = DI;
      g.R = mem; g.ldr = DM; g.Cf = mem; g.ldc = DM; g.Kp = DI;
      launch_gemm(stream, g, BW1, DM); }
  }
  cvt(mem, memh, BW1, BW1, DM, DM);

  // ---- Transformer decoder ----
  k_embed_dec<<<cdiv(BATCH * SEQ2 * DM, 256), 256, 0, stream>>>(yin, d_emb_w, d_emb_b, xdec, xh);
  const int ROWS = BATCH * SEQ2;

  for (int l = 0; l < NLAYER; l++) {
    const _Float16* saw = sainh + (size_t)l * 3 * DM * DM;
    const float*    sab = sa_in_b + l * 3 * DM;
    const _Float16* caw = cainh + (size_t)l * 3 * DM * DM;
    const float*    cab = ca_in_b + l * 3 * DM;

    // ===== self-attention =====
    { GemmArgs g{}; g.A = xh; g.lda = DM; g.W = saw; g.ldw = DM; g.bias = sab;
      g.Ch = qh; g.ldch = DM; g.Kp = DM; launch_gemm(stream, g, ROWS, DM); }
    { GemmArgs g{}; g.A = xh; g.lda = DM; g.W = saw + (size_t)DM * DM; g.ldw = DM;
      g.bias = sab + DM; g.Ch = kh; g.ldch = DM; g.Kp = DM; launch_gemm(stream, g, ROWS, DM); }
    { GemmArgs g{}; g.A = xh; g.lda = DM; g.sAb = SEQ2 * DM;
      g.W = saw + (size_t)2 * DM * DM; g.ldw = DM; g.bias = sab + 2 * DM;
      g.Ct = VtS; g.ldct = 128; g.sCtb = (long long)DM * 128;
      g.Kp = DM; g.Bb = BATCH; launch_gemm(stream, g, SEQ2, DM); }
    { GemmArgs g{}; g.A = qh; g.lda = DM; g.sAb = SEQ2 * DM; g.sAh = DHEAD;
      g.W = kh; g.ldw = DM; g.sWb = SEQ2 * DM; g.sWh = DHEAD;
      g.Cf = scoresS; g.ldc = 112; g.sCb = (long long)NHEAD * 112 * 112; g.sCh = 112 * 112;
      g.Kp = DHEAD; g.Bb = BATCH; g.Hh = NHEAD; g.scale = ATTN_SCALE;
      launch_gemm(stream, g, SEQ2, SEQ2); }
    k_softmax<<<BATCH * NHEAD * SEQ2, 128, 0, stream>>>(scoresS, probsS, SEQ2, SEQ2,
        112, 112 * 112, 128, 112 * 128, 128);
    { GemmArgs g{}; g.A = probsS; g.lda = 128;
      g.sAb = (long long)NHEAD * 112 * 128; g.sAh = 112 * 128;
      g.W = VtS; g.ldw = 128; g.sWb = (long long)DM * 128; g.sWh = (long long)DHEAD * 128;
      g.Ch = attnh; g.ldch = DM; g.sChb = SEQ2 * DM; g.sChh = DHEAD;
      g.Kp = 128; g.Bb = BATCH; g.Hh = NHEAD; launch_gemm(stream, g, SEQ2, DHEAD); }
    { GemmArgs g{}; g.A = attnh; g.lda = DM; g.W = saouth + (size_t)l * DM * DM; g.ldw = DM;
      g.bias = sa_out_b + l * DM; g.R = xdec; g.ldr = DM; g.Cf = tmp; g.ldc = DM;
      g.Kp = DM; launch_gemm(stream, g, ROWS, DM); }
    k_layernorm<<<ROWS, DM, 0, stream>>>(tmp, ln1_w + l * DM, ln1_b + l * DM, xdec, xh);

    // ===== cross-attention =====
    { GemmArgs g{}; g.A = xh; g.lda = DM; g.W = caw; g.ldw = DM; g.bias = cab;
      g.Ch = qh; g.ldch = DM; g.Kp = DM; launch_gemm(stream, g, ROWS, DM); }
    { GemmArgs g{}; g.A = memh; g.lda = DM; g.W = caw + (size_t)DM * DM; g.ldw = DM;
      g.bias = cab + DM; g.Ch = khc; g.ldch = DM; g.Kp = DM; launch_gemm(stream, g, BW1, DM); }
    { GemmArgs g{}; g.A = memh; g.lda = DM; g.sAb = (long long)SEQ1 * DM;
      g.W = caw + (size_t)2 * DM * DM; g.ldw = DM; g.bias = cab + 2 * DM;
      g.Ct = Vtc; g.ldct = SEQ1; g.sCtb = (long long)DM * SEQ1;
      g.Kp = DM; g.Bb = BATCH; launch_gemm(stream, g, SEQ1, DM); }
    { GemmArgs g{}; g.A = qh; g.lda = DM; g.sAb = SEQ2 * DM; g.sAh = DHEAD;
      g.W = khc; g.ldw = DM; g.sWb = (long long)SEQ1 * DM; g.sWh = DHEAD;
      g.Cf = scoresC; g.ldc = SEQ1; g.sCb = (long long)NHEAD * 112 * SEQ1;
      g.sCh = (long long)112 * SEQ1;
      g.Kp = DHEAD; g.Bb = BATCH; g.Hh = NHEAD; g.scale = ATTN_SCALE;
      launch_gemm(stream, g, SEQ2, SEQ1); }
    k_softmax<<<BATCH * NHEAD * SEQ2, 128, 0, stream>>>(scoresC, probsC, SEQ2, SEQ1,
        SEQ1, (long long)112 * SEQ1, SEQ1, (long long)112 * SEQ1, SEQ1);
    { GemmArgs g{}; g.A = probsC; g.lda = SEQ1;
      g.sAb = (long long)NHEAD * 112 * SEQ1; g.sAh = (long long)112 * SEQ1;
      g.W = Vtc; g.ldw = SEQ1; g.sWb = (long long)DM * SEQ1; g.sWh = (long long)DHEAD * SEQ1;
      g.Ch = attnh; g.ldch = DM; g.sChb = SEQ2 * DM; g.sChh = DHEAD;
      g.Kp = SEQ1; g.Bb = BATCH; g.Hh = NHEAD; launch_gemm(stream, g, SEQ2, DHEAD); }
    { GemmArgs g{}; g.A = attnh; g.lda = DM; g.W = caouth + (size_t)l * DM * DM; g.ldw = DM;
      g.bias = ca_out_b + l * DM; g.R = xdec; g.ldr = DM; g.Cf = tmp; g.ldc = DM;
      g.Kp = DM; launch_gemm(stream, g, ROWS, DM); }
    k_layernorm<<<ROWS, DM, 0, stream>>>(tmp, ln2_w + l * DM, ln2_b + l * DM, xdec, xh);

    // ===== FFN =====
    { GemmArgs g{}; g.A = xh; g.lda = DM; g.W = ffw1h + (size_t)l * DFF * DM; g.ldw = DM;
      g.bias = ff_b1 + l * DFF; g.Ch = ffh; g.ldch = DFF; g.Kp = DM; g.relu = 1;
      launch_gemm(stream, g, ROWS, DFF); }
    { GemmArgs g{}; g.A = ffh; g.lda = DFF; g.W = ffw2h + (size_t)l * DM * DFF; g.ldw = DFF;
      g.bias = ff_b2 + l * DM; g.R = xdec; g.ldr = DM; g.Cf = tmp; g.ldc = DM; g.Kp = DFF;
      launch_gemm(stream, g, ROWS, DM); }
    k_layernorm<<<ROWS, DM, 0, stream>>>(tmp, ln3_w + l * DM, ln3_b + l * DM, xdec, xh);
  }

  k_layernorm<<<ROWS, DM, 0, stream>>>(xdec, fn_w, fn_b, xf, nullptr);
  k_extract<<<cdiv(ROWS, 256), 256, 0, stream>>>(xf, out);
}
